// SegNetViT_89953795047603
// MI455X (gfx1250) — matrix operand order repaired and checked by reading, not compiled, not run on hardware
//
#include <hip/hip_runtime.h>
#include <hip/hip_bf16.h>

// ---------------------------------------------------------------------------
// SegNetViT (LightGlue-style) forward for MI455X / gfx1250.
// All GEMMs run on v_wmma_f32_16x16x32_bf16 (fp32 accumulate).
// All operands pre-converted to bf16 once; GEMM inner loops are pure b128
// loads + WMMA. Attention is fused flash-style with a lane-local softmax:
// scores take one f32 LDS transpose, so row reductions are local VALU plus a
// single xor-16 shuffle (no ds_bpermute butterflies on the critical path).
// ---------------------------------------------------------------------------

typedef __attribute__((ext_vector_type(16))) __bf16          v16bf;
typedef __attribute__((ext_vector_type(8)))  float           v8f;
typedef __attribute__((ext_vector_type(16))) unsigned short  v16u;
typedef __attribute__((ext_vector_type(8)))  unsigned short  v8u;

#define BATCH  4
#define SEQ    2048
#define DIM    256
#define HEADS  4
#define HDIM   64
#define LAYERS 9
#define MROWS  (BATCH * SEQ)   // 8192 token rows

// fp32 -> bf16, round-to-nearest-even
__device__ __forceinline__ unsigned short f2bf(float f) {
  unsigned int u = __float_as_uint(f);
  u += 0x7FFFu + ((u >> 16) & 1u);
  return (unsigned short)(u >> 16);
}

__device__ __forceinline__ v16bf bf_from_u16(v16u t) {
  return __builtin_bit_cast(v16bf, t);
}
__device__ __forceinline__ v16u cat8(v8u a, v8u b) {
  return __builtin_shufflevector(a, b, 0,1,2,3,4,5,6,7,8,9,10,11,12,13,14,15);
}
// A-operand (16-bit A 16x32 layout): lane m holds K = kb+8*half+e, kb+16+8*half+e
__device__ __forceinline__ v16bf a_bf16(const unsigned short* p) {
  return bf_from_u16(cat8(*(const v8u*)(p), *(const v8u*)(p + 16)));
}
// B-operand: lane n holds K = kb+8*half+e, kb+16+8*half+e (same order as A)
__device__ __forceinline__ v16bf b_bf16(const unsigned short* p) {
  return bf_from_u16(cat8(*(const v8u*)(p), *(const v8u*)(p + 16)));
}

#define WMMA_BF16(a, b, c) \
  __builtin_amdgcn_wmma_f32_16x16x32_bf16(false, (a), false, (b), (short)0, (c), false, false)

// ---------------------------------------------------------------------------
// fp32 -> bf16 buffer conversion (vectorized, one shot per tensor)
// ---------------------------------------------------------------------------
__global__ void cvt_f32_bf16_kernel(const float* __restrict__ in,
                                    unsigned short* __restrict__ out, int n)
{
  int i = (blockIdx.x * blockDim.x + threadIdx.x) * 4;
  if (i >= n) return;
  float4 v = *(const float4*)(in + i);
  ushort4 o;
  o.x = f2bf(v.x); o.y = f2bf(v.y); o.z = f2bf(v.z); o.w = f2bf(v.w);
  *(ushort4*)(out + i) = o;
}

// ---------------------------------------------------------------------------
// GEMM: Y[M,Nout] = concatK(A1[:, :KSPLIT], A2[:, KSPLIT:K]) @ W^T + bias (+res)
// A1/A2/W are bf16; W is [Nout, K] row-major (torch Linear weight).
// 256 threads = 8 waves; block tile 128(M) x 32(N); each wave owns two 16x16
// D tiles (rows +0 / +64) sharing one B operand -> 2 WMMA per B load.
// Optional fp32 output Y and/or bf16 mirror Ybf.
// ---------------------------------------------------------------------------
template <int K, int KSPLIT>
__global__ __launch_bounds__(256) void gemm_bf16_wmma(
    const unsigned short* __restrict__ A1, const unsigned short* __restrict__ A2,
    const unsigned short* __restrict__ W, const float* __restrict__ bias,
    const float* __restrict__ res, float* __restrict__ Y,
    unsigned short* __restrict__ Ybf, int Nout)
{
  const int lane = threadIdx.x & 31;
  const int wave = threadIdx.x >> 5;
  const int half = lane >> 4;
  const int lq   = lane & 15;
  const int rowBase = blockIdx.x * 128 + (wave >> 1) * 16;   // tile0; tile1 = +64
  const int colBase = blockIdx.y * 32 + (wave & 1) * 16;
  const int am = rowBase + lq;   // A row fed by this lane (tile0)
  const int bn = colBase + lq;   // output column fed/owned by this lane

  const unsigned short* wrow = W + (size_t)bn * K + 8 * half;
  const unsigned short* a1r0 = A1 + (size_t)am * KSPLIT + 8 * half;
  const unsigned short* a1r1 = a1r0 + (size_t)64 * KSPLIT;

  v8f acc0 = {0.f,0.f,0.f,0.f,0.f,0.f,0.f,0.f};
  v8f acc1 = {0.f,0.f,0.f,0.f,0.f,0.f,0.f,0.f};

#pragma unroll 8
  for (int kb = 0; kb < KSPLIT; kb += 32) {
    v16bf b  = b_bf16(wrow + kb);
    acc0 = WMMA_BF16(a_bf16(a1r0 + kb), b, acc0);
    acc1 = WMMA_BF16(a_bf16(a1r1 + kb), b, acc1);
  }
  if constexpr (KSPLIT < K) {
    const unsigned short* a2r0 = A2 + (size_t)am * (K - KSPLIT) + 8 * half;
    const unsigned short* a2r1 = a2r0 + (size_t)64 * (K - KSPLIT);
#pragma unroll 8
    for (int kb = 0; kb < K - KSPLIT; kb += 32) {
      v16bf b  = b_bf16(wrow + KSPLIT + kb);
      acc0 = WMMA_BF16(a_bf16(a2r0 + kb), b, acc0);
      acc1 = WMMA_BF16(a_bf16(a2r1 + kb), b, acc1);
    }
  }

  // epilogue: C/D layout row = r + 8*half; all guards are wave-uniform
  const float bv = bias[bn];
  const int base0 = (rowBase + 8 * half) * Nout + bn;      // < 2^31
  const int base1 = base0 + 64 * Nout;
  float o0[8], o1[8];
#pragma unroll
  for (int r = 0; r < 8; ++r) { o0[r] = acc0[r] + bv; o1[r] = acc1[r] + bv; }
  if (res) {
#pragma unroll
    for (int r = 0; r < 8; ++r) {
      o0[r] += res[base0 + r * Nout];
      o1[r] += res[base1 + r * Nout];
    }
  }
  if (Y) {
#pragma unroll
    for (int r = 0; r < 8; ++r) {
      Y[base0 + r * Nout] = o0[r];
      Y[base1 + r * Nout] = o1[r];
    }
  }
  if (Ybf) {
#pragma unroll
    for (int r = 0; r < 8; ++r) {
      Ybf[base0 + r * Nout] = f2bf(o0[r]);
      Ybf[base1 + r * Nout] = f2bf(o1[r]);
    }
  }
}

// ---------------------------------------------------------------------------
// Fourier positional encoder: proj = kp @ Wr^T (32 freqs), cache cos/sin.
// ---------------------------------------------------------------------------
__global__ void posenc_kernel(const float* __restrict__ kp, const float* __restrict__ Wr,
                              float* __restrict__ cs, float* __restrict__ sn)
{
  int tid = blockIdx.x * blockDim.x + threadIdx.x;
  if (tid >= MROWS * 32) return;
  int f = tid & 31, bn = tid >> 5;
  float d = kp[bn * 2 + 0] * Wr[f * 2 + 0] + kp[bn * 2 + 1] * Wr[f * 2 + 1];
  cs[tid] = __cosf(d);
  sn[tid] = __sinf(d);
}

// ---------------------------------------------------------------------------
// Extract q,k,v from interleaved qkv (o = h*192 + d*3 + c), apply RoPE to q,k,
// emit bf16. The softmax scale 1/sqrt(64) is folded into q here.
// q,k -> [BH, N, 64]; v -> transposed [BH, 64, N] so the PV B-operand is
// contiguous along keys.
// ---------------------------------------------------------------------------
__global__ void rope_extract_kernel(const float* __restrict__ qkv,
    const float* __restrict__ cs, const float* __restrict__ sn,
    unsigned short* __restrict__ qr, unsigned short* __restrict__ kr,
    unsigned short* __restrict__ vT)
{
  int tid = blockIdx.x * blockDim.x + threadIdx.x;     // BH * SEQ * 32
  if (tid >= BATCH * HEADS * SEQ * 32) return;
  int i  = tid & 31;
  int n  = (tid >> 5) & (SEQ - 1);
  int bh = tid >> 16;
  int b = bh >> 2, h = bh & 3;
  size_t base = ((size_t)(b * SEQ + n)) * (3 * DIM) + h * 192 + i * 6;
  float q0 = qkv[base + 0], k0 = qkv[base + 1], v0 = qkv[base + 2];
  float q1 = qkv[base + 3], k1 = qkv[base + 4], v1 = qkv[base + 5];
  float c = cs[(b * SEQ + n) * 32 + i];
  float s = sn[(b * SEQ + n) * 32 + i];
  size_t ro = ((size_t)(bh * SEQ + n)) * HDIM + 2 * i;
  qr[ro + 0] = f2bf((q0 * c - q1 * s) * 0.125f);   // softmax scale folded in
  qr[ro + 1] = f2bf((q1 * c + q0 * s) * 0.125f);
  kr[ro + 0] = f2bf(k0 * c - k1 * s);
  kr[ro + 1] = f2bf(k1 * c + k0 * s);
  size_t vo = ((size_t)(bh * HDIM + 2 * i)) * SEQ + n;
  vT[vo]       = f2bf(v0);
  vT[vo + SEQ] = f2bf(v1);
}

// ---------------------------------------------------------------------------
// Fused flash attention (bf16 WMMA, fp32 accumulate, online softmax).
// Wave = 16 query rows; block = 8 waves = 128 queries. Per 32-key step:
// 4 WMMA (S = q@k^T), f32 LDS transpose of S, lane-local softmax (row = lane),
// P packed directly into the A-operand, 4 WMMA (acc += P@V).
// ---------------------------------------------------------------------------
__global__ __launch_bounds__(256) void attn_kernel(
    const unsigned short* __restrict__ qr, const unsigned short* __restrict__ kr,
    const unsigned short* __restrict__ vT, unsigned short* __restrict__ msg)
{
  __shared__ __align__(64) float sl[8][16][32];  // per-wave raw score tile
  __shared__ __align__(64) float bl[8][16];      // per-wave row-stat broadcast
  const int lane = threadIdx.x & 31;
  const int wave = threadIdx.x >> 5;
  const int half = lane >> 4;
  const int lq   = lane & 15;
  const int bh   = blockIdx.x;
  const int b = bh >> 2, h = bh & 3;
  const int q0row = blockIdx.y * 128 + wave * 16;

  const unsigned short* qp = qr + ((size_t)(bh * SEQ + q0row + lq)) * HDIM;
  v16bf qa0 = a_bf16(qp +      8 * half);   // d 0..31
  v16bf qa1 = a_bf16(qp + 32 + 8 * half);   // d 32..63

  v8f acc0 = {}, acc1 = {}, acc2 = {}, acc3 = {};
  float m_row = -1e30f, l_row = 0.f;  // stats for row lq (duplicated per half)

  const unsigned short* kbase = kr + (size_t)bh * SEQ * HDIM;
  const unsigned short* vbase = vT + (size_t)bh * HDIM * SEQ;

  for (int j = 0; j < SEQ; j += 32) {
    const unsigned short* k0p = kbase + (size_t)(j + lq) * HDIM + 8 * half;
    const unsigned short* k1p = k0p + 16 * HDIM;
    v16bf b00 = b_bf16(k0p);
    v16bf b01 = b_bf16(k0p + 32);
    v16bf b10 = b_bf16(k1p);
    v16bf b11 = b_bf16(k1p + 32);

    v8f z = {};
    v8f S0 = WMMA_BF16(qa0, b00, z);
    S0     = WMMA_BF16(qa1, b01, S0);
    v8f S1 = WMMA_BF16(qa0, b10, z);
    S1     = WMMA_BF16(qa1, b11, S1);

    // C layout -> LDS (f32); lane owns rows r+8*half, cols lq / 16+lq
#pragma unroll
    for (int r = 0; r < 8; ++r) {
      sl[wave][r + 8 * half][lq]      = S0[r];
      sl[wave][r + 8 * half][16 + lq] = S1[r];
    }
    asm volatile("s_wait_dscnt 0" ::: "memory");
    // lane reads its 16 values of row lq in A-operand order
    float4 p0 = *(const float4*)&sl[wave][lq][8 * half];
    float4 p1 = *(const float4*)&sl[wave][lq][8 * half + 4];
    float4 p2 = *(const float4*)&sl[wave][lq][16 + 8 * half];
    float4 p3 = *(const float4*)&sl[wave][lq][16 + 8 * half + 4];
    float v[16] = { p0.x,p0.y,p0.z,p0.w, p1.x,p1.y,p1.z,p1.w,
                    p2.x,p2.y,p2.z,p2.w, p3.x,p3.y,p3.z,p3.w };
    float mx = v[0];
#pragma unroll
    for (int i = 1; i < 16; ++i) mx = fmaxf(mx, v[i]);
    mx = fmaxf(mx, __shfl_xor(mx, 16, 32));       // merge the two column halves
    float mn = fmaxf(m_row, mx);
    float alpha = __expf(m_row - mn);
    m_row = mn;
    float sm = 0.f;
    v16u t;
#pragma unroll
    for (int i = 0; i < 16; ++i) {
      float e = __expf(v[i] - mn);
      sm += e;
      t[i] = f2bf(e);
    }
    v16bf pa = bf_from_u16(t);                    // P already in A-operand order
    sm += __shfl_xor(sm, 16, 32);
    l_row = l_row * alpha + sm;

    // broadcast alpha (indexed by row) to the C-layout row owners
    bl[wave][lq] = alpha;                         // halves write identical value
    asm volatile("s_wait_dscnt 0" ::: "memory");
    float4 a0 = *(const float4*)&bl[wave][8 * half];
    float4 a1 = *(const float4*)&bl[wave][8 * half + 4];
    float aC[8] = { a0.x,a0.y,a0.z,a0.w, a1.x,a1.y,a1.z,a1.w };
#pragma unroll
    for (int r = 0; r < 8; ++r) {
      acc0[r] *= aC[r]; acc1[r] *= aC[r]; acc2[r] *= aC[r]; acc3[r] *= aC[r];
    }

    const unsigned short* vp = vbase + (size_t)lq * SEQ + j + 8 * half;
    acc0 = WMMA_BF16(pa, b_bf16(vp),            acc0);
    acc1 = WMMA_BF16(pa, b_bf16(vp + 16 * SEQ), acc1);
    acc2 = WMMA_BF16(pa, b_bf16(vp + 32 * SEQ), acc2);
    acc3 = WMMA_BF16(pa, b_bf16(vp + 48 * SEQ), acc3);
  }

  // broadcast 1/l and write msg[B, N, 256] as bf16 (heads re-interleaved)
  bl[wave][lq] = 1.0f / l_row;
  asm volatile("s_wait_dscnt 0" ::: "memory");
  float4 a0 = *(const float4*)&bl[wave][8 * half];
  float4 a1 = *(const float4*)&bl[wave][8 * half + 4];
  float iC[8] = { a0.x,a0.y,a0.z,a0.w, a1.x,a1.y,a1.z,a1.w };
#pragma unroll
  for (int r = 0; r < 8; ++r) {
    int row = b * SEQ + q0row + r + 8 * half;
    unsigned short* mp = msg + (size_t)row * DIM + h * HDIM + lq;
    mp[0]  = f2bf(acc0[r] * iC[r]);
    mp[16] = f2bf(acc1[r] * iC[r]);
    mp[32] = f2bf(acc2[r] * iC[r]);
    mp[48] = f2bf(acc3[r] * iC[r]);
  }
}

// ---------------------------------------------------------------------------
// Row LayerNorm + exact GELU: fp32 in, bf16 out. One wave per row.
// ---------------------------------------------------------------------------
__global__ __launch_bounds__(256) void ln_gelu_kernel(const float* __restrict__ X,
    unsigned short* __restrict__ O,
    const float* __restrict__ g, const float* __restrict__ bta, int F)
{
  int wave = threadIdx.x >> 5, lane = threadIdx.x & 31;
  int row = blockIdx.x * 8 + wave;
  const float* xr = X + (size_t)row * F;
  unsigned short* orow = O + (size_t)row * F;
  float s = 0.f, s2 = 0.f;
  for (int j = lane; j < F; j += 32) { float v = xr[j]; s += v; s2 += v * v; }
#pragma unroll
  for (int off = 1; off < 32; off <<= 1) {
    s  += __shfl_xor(s,  off, 32);
    s2 += __shfl_xor(s2, off, 32);
  }
  float mu  = s / F;
  float var = s2 / F - mu * mu;
  float rs  = rsqrtf(var + 1e-5f);
  for (int j = lane; j < F; j += 32) {
    float v = (xr[j] - mu) * rs * g[j] + bta[j];
    orow[j] = f2bf(0.5f * v * (1.0f + erff(v * 0.70710678118f)));
  }
}

// ---------------------------------------------------------------------------
extern "C" void kernel_launch(void* const* d_in, const int* in_sizes, int n_in,
                              void* d_out, int out_size, void* d_ws, size_t ws_size,
                              hipStream_t stream)
{
  const float* descriptors = (const float*)d_in[0];
  const float* kp      = (const float*)d_in[1];
  const float* Wr      = (const float*)d_in[2];
  const float* Wp      = (const float*)d_in[3];
  const float* bp      = (const float*)d_in[4];
  const float* qkv_W   = (const float*)d_in[5];
  const float* qkv_b   = (const float*)d_in[6];
  const float* proj_W  = (const float*)d_in[7];
  const float* proj_b  = (const float*)d_in[8];
  const float* mlp_W1  = (const float*)d_in[9];
  const float* mlp_b1  = (const float*)d_in[10];
  const float* ln_g    = (const float*)d_in[11];
  const float* ln_b    = (const float*)d_in[12];
  const float* mlp_W2  = (const float*)d_in[13];
  const float* mlp_b2  = (const float*)d_in[14];
  const float* seg_W1  = (const float*)d_in[15];
  const float* seg_b1  = (const float*)d_in[16];
  const float* seg_g   = (const float*)d_in[17];
  const float* seg_bb  = (const float*)d_in[18];
  const float* seg_W2  = (const float*)d_in[19];
  const float* seg_b2  = (const float*)d_in[20];
  (void)in_sizes; (void)n_in; (void)out_size; (void)ws_size;

  char* ws = (char*)d_ws;
  size_t off = 0;
  auto alloc = [&](size_t bytes) -> void* {
    void* p = ws + off;
    off += (bytes + 255) & ~(size_t)255;
    return p;
  };
  // fp32 buffers
  float* x    = (float*)alloc((size_t)MROWS * DIM * 4);
  float* cs   = (float*)alloc((size_t)MROWS * 32 * 4);
  float* sn   = (float*)alloc((size_t)MROWS * 32 * 4);
  float* qkv  = (float*)alloc((size_t)MROWS * 768 * 4);
  float* hbuf = (float*)alloc((size_t)MROWS * 512 * 4);
  float* y1   = (float*)alloc((size_t)MROWS * 1024 * 4);
  // bf16 activation buffers
  unsigned short* dbf    = (unsigned short*)alloc((size_t)MROWS * DIM * 2);
  unsigned short* xbf    = (unsigned short*)alloc((size_t)MROWS * DIM * 2);
  unsigned short* qr     = (unsigned short*)alloc((size_t)16 * SEQ * HDIM * 2);
  unsigned short* kr     = (unsigned short*)alloc((size_t)16 * SEQ * HDIM * 2);
  unsigned short* vT     = (unsigned short*)alloc((size_t)16 * HDIM * SEQ * 2);
  unsigned short* msgbf  = (unsigned short*)alloc((size_t)MROWS * DIM * 2);
  unsigned short* msgpbf = (unsigned short*)alloc((size_t)MROWS * DIM * 2);
  unsigned short* hbf    = (unsigned short*)alloc((size_t)MROWS * 512 * 2);
  unsigned short* y1bf   = (unsigned short*)alloc((size_t)MROWS * 1024 * 2);
  // bf16 weights (converted once per launch)
  unsigned short* wp_bf  = (unsigned short*)alloc((size_t)DIM * DIM * 2);
  unsigned short* wqkv   = (unsigned short*)alloc((size_t)LAYERS * 768 * DIM * 2);
  unsigned short* wproj  = (unsigned short*)alloc((size_t)LAYERS * DIM * DIM * 2);
  unsigned short* wm1    = (unsigned short*)alloc((size_t)LAYERS * 512 * 512 * 2);
  unsigned short* wm2    = (unsigned short*)alloc((size_t)LAYERS * DIM * 512 * 2);
  unsigned short* ws1    = (unsigned short*)alloc((size_t)1024 * DIM * 2);
  unsigned short* ws2    = (unsigned short*)alloc((size_t)512 * 1024 * 2);

  auto cvt = [&](const float* src, unsigned short* dst, int n) {
    cvt_f32_bf16_kernel<<<(n / 4 + 255) / 256, 256, 0, stream>>>(src, dst, n);
  };
  // one-shot conversions
  cvt(descriptors, dbf, MROWS * DIM);
  cvt(Wp,     wp_bf, DIM * DIM);
  cvt(qkv_W,  wqkv,  LAYERS * 768 * DIM);
  cvt(proj_W, wproj, LAYERS * DIM * DIM);
  cvt(mlp_W1, wm1,   LAYERS * 512 * 512);
  cvt(mlp_W2, wm2,   LAYERS * DIM * 512);
  cvt(seg_W1, ws1,   1024 * DIM);
  cvt(seg_W2, ws2,   512 * 1024);

  posenc_kernel<<<(MROWS * 32) / 256, 256, 0, stream>>>(kp, Wr, cs, sn);

  // input projection: x = descriptors @ Wp^T + bp (fp32 + bf16 mirror)
  gemm_bf16_wmma<256, 256><<<dim3(MROWS / 128, DIM / 32), 256, 0, stream>>>(
      dbf, nullptr, wp_bf, bp, nullptr, x, xbf, DIM);

  for (int l = 0; l < LAYERS; ++l) {
    gemm_bf16_wmma<256, 256><<<dim3(MROWS / 128, 768 / 32), 256, 0, stream>>>(
        xbf, nullptr, wqkv + (size_t)l * 768 * DIM, qkv_b + l * 768,
        nullptr, qkv, nullptr, 768);
    rope_extract_kernel<<<(16 * SEQ * 32) / 256, 256, 0, stream>>>(
        qkv, cs, sn, qr, kr, vT);
    attn_kernel<<<dim3(16, SEQ / 128), 256, 0, stream>>>(qr, kr, vT, msgbf);
    gemm_bf16_wmma<256, 256><<<dim3(MROWS / 128, DIM / 32), 256, 0, stream>>>(
        msgbf, nullptr, wproj + (size_t)l * DIM * DIM, proj_b + l * DIM,
        nullptr, nullptr, msgpbf, DIM);
    // MLP1 on concat([x, msgp]) via compile-time K-split dual-A GEMM
    gemm_bf16_wmma<512, 256><<<dim3(MROWS / 128, 512 / 32), 256, 0, stream>>>(
        xbf, msgpbf, wm1 + (size_t)l * 512 * 512, mlp_b1 + l * 512,
        nullptr, hbuf, nullptr, 512);
    ln_gelu_kernel<<<MROWS / 8, 256, 0, stream>>>(
        hbuf, hbf, ln_g + l * 512, ln_b + l * 512, 512);
    // MLP2 with fused residual; refreshes both x (fp32) and xbf (bf16)
    gemm_bf16_wmma<512, 512><<<dim3(MROWS / 128, DIM / 32), 256, 0, stream>>>(
        hbf, nullptr, wm2 + (size_t)l * DIM * 512, mlp_b2 + l * DIM,
        x, x, xbf, DIM);
  }

  // seg head: 256 -> 1024 -> LN+GELU -> 512
  gemm_bf16_wmma<256, 256><<<dim3(MROWS / 128, 1024 / 32), 256, 0, stream>>>(
      xbf, nullptr, ws1, seg_b1, nullptr, y1, nullptr, 1024);
  ln_gelu_kernel<<<MROWS / 8, 256, 0, stream>>>(y1, y1bf, seg_g, seg_bb, 1024);
  gemm_bf16_wmma<1024, 1024><<<dim3(MROWS / 128, 512 / 32), 256, 0, stream>>>(
      y1bf, nullptr, ws2, seg_b2, nullptr, (float*)d_out, nullptr, 512);
}
